// MultiHeadAttention_61289183314138
// MI455X (gfx1250) — matrix operand order repaired and checked by reading, not compiled, not run on hardware
//
#include <hip/hip_runtime.h>
#include <hip/hip_bf16.h>

// Shapes from reference: B=2, S=2048, D=1024, H=16, DH=64
#define BB 2
#define SS 2048
#define DD 1024
#define HH 16
#define DHH 64

typedef __attribute__((ext_vector_type(16))) _Float16 v16h;
typedef __attribute__((ext_vector_type(8)))  _Float16 v8h;
typedef __attribute__((ext_vector_type(8)))  float    v8f;
typedef __attribute__((ext_vector_type(4)))  float    v4f;
typedef __attribute__((ext_vector_type(4)))  unsigned int u32x4;
typedef __attribute__((ext_vector_type(8)))  int      i32x8;
typedef __attribute__((ext_vector_type(4)))  int      i32x4;

#if defined(__HIP_DEVICE_COMPILE__) && __has_builtin(__builtin_amdgcn_tensor_load_to_lds)
#define HAVE_TDM 1
#else
#define HAVE_TDM 0
#endif

// Build a 16-half fragment from two contiguous 8-half chunks.
__device__ __forceinline__ v16h make_frag(const _Float16* lo, const _Float16* hi) {
  v16h r;
  v8h a = *(const v8h*)lo;
  v8h b = *(const v8h*)hi;
#pragma unroll
  for (int i = 0; i < 8; ++i) { r[i] = a[i]; r[i + 8] = b[i]; }
  return r;
}

__device__ __forceinline__ v8f wmma_f16(v16h a, v16h b, v8f c) {
  return __builtin_amdgcn_wmma_f32_16x16x32_f16(false, a, false, b, (short)0, c,
                                                false, false);
}

#if HAVE_TDM
// Issue a TDM 2D tile load: 64 rows x 32 cols of f16 from a row-major tensor
// with row stride 1024 elements, into LDS at byte offset lds_off.
// D# packing per CDNA5 ISA ch.8 (group0 128b, group1 256b; groups 2/3 zero => 2D).
__device__ __forceinline__ void tdm_load_tile(const _Float16* gsrc, unsigned lds_off,
                                              int rows_total) {
  unsigned long long ga = (unsigned long long)(size_t)gsrc;
  u32x4 g0;
  g0[0] = 1u;                                          // count=1 (valid), no gather
  g0[1] = lds_off;                                     // lds_addr (bytes)
  g0[2] = (unsigned)ga;                                // global_addr[31:0]
  g0[3] = (unsigned)((ga >> 32) & 0x1FFFFFFu) | (2u << 30); // addr[56:32] | type=2
  i32x8 g1;
  g1[0] = (1 << 16);                                   // data_size = 2 bytes
  g1[1] = (int)(1024u << 16);                          // tensor_dim0 = 1024 (lo16)
  g1[2] = (int)((unsigned)rows_total << 16);           // dim0 hi=0 | tensor_dim1 lo16
  g1[3] = (32 << 16);                                  // dim1 hi=0 | tile_dim0 = 32
  g1[4] = 64;                                          // tile_dim1 = 64, tile_dim2 = 0
  g1[5] = 1024;                                        // tensor_dim0_stride lo32
  g1[6] = 0;
  g1[7] = 0;
  i32x4 z4 = {0, 0, 0, 0};
#if __clang_major__ >= 23
  i32x8 z8 = {0, 0, 0, 0, 0, 0, 0, 0};
  __builtin_amdgcn_tensor_load_to_lds(g0, g1, z4, z4, z8, 0);
#else
  __builtin_amdgcn_tensor_load_to_lds(g0, g1, z4, z4, 0);
#endif
}
#endif

// ---------------------------------------------------------------------------
// cvt kernel: f32 -> f16 elementwise (n multiple of 8)
// ---------------------------------------------------------------------------
__global__ void cvt_f32_to_f16(const float* __restrict__ src, _Float16* __restrict__ dst,
                               int n) {
  const int i = (blockIdx.x * blockDim.x + threadIdx.x) * 8;
  if (i + 8 <= n) {
    v4f a = *(const v4f*)(src + i);
    v4f b = *(const v4f*)(src + i + 4);
    v8h o;
#pragma unroll
    for (int j = 0; j < 4; ++j) { o[j] = (_Float16)a[j]; o[j + 4] = (_Float16)b[j]; }
    *(v8h*)(dst + i) = o;
  }
}

// ---------------------------------------------------------------------------
// gemm16_kernel: Y = A(f16, Mx1024) @ W^T(f16, Nx1024) + bias
// out_mode 0: f16 out16[((b*H+h)*S + s)*DH + dh]   (Q, K per-head)
// out_mode 1: f16 out16[((b*H+h)*DH + dh)*S + s]   (V transposed)
// out_mode 2: f32 out32[m*1024 + n]                (final projection)
// 64x64 tile, 256 threads (8 waves), TDM double-buffered LDS staging.
// ---------------------------------------------------------------------------
__global__ void gemm16_kernel(const _Float16* __restrict__ A, const _Float16* __restrict__ W,
                              const float* __restrict__ bias,
                              _Float16* __restrict__ out16, float* __restrict__ out32,
                              int out_mode) {
  constexpr int K = DD;
  constexpr int KSTEPS = K / 32;
  // [buf][A|B][64*32] halves; single LDS object => base offset 0.
  __shared__ __align__(16) _Float16 tile[4 * 64 * 32];

  const int m0 = blockIdx.x * 64;
  const int n0 = blockIdx.y * 64;
  const int tid = threadIdx.x;
  const int lane = tid & 31;
  const int wave = tid >> 5;
  const int mi  = wave >> 1;          // 0..3
  const int ni0 = (wave & 1) * 2;     // 0 or 2
  const int row = lane & 15;
  const int hl  = lane >> 4;          // 0 or 1

  v8f acc0 = {}; v8f acc1 = {};

#if HAVE_TDM
  if (wave == 0) {
    tdm_load_tile(A + (size_t)m0 * K, 0u,    BB * SS);   // A tile -> buf0
    tdm_load_tile(W + (size_t)n0 * K, 4096u, DD);        // B tile -> buf0
  }
#endif

  for (int i = 0; i < KSTEPS; ++i) {
    const int kk = i * 32;
#if HAVE_TDM
    const int ib = i & 1;
    if (wave == 0) {
      if (i + 1 < KSTEPS) {
        const unsigned nb = (unsigned)((i + 1) & 1) * 8192u;
        tdm_load_tile(A + (size_t)m0 * K + kk + 32, nb,         BB * SS);
        tdm_load_tile(W + (size_t)n0 * K + kk + 32, nb + 4096u, DD);
        __builtin_amdgcn_s_wait_tensorcnt(2);   // pair i complete (in-order)
      } else {
        __builtin_amdgcn_s_wait_tensorcnt(0);
      }
    }
    __syncthreads();
    // LDS is written only by the TDM engine. Escape the tile address as an
    // asm *input* with a memory clobber: the compiler must assume the asm
    // writes the tile (so fragment loads cannot fold to undef), while the
    // loads below still go through `tile` directly and keep address space 3
    // (ds_load, not flat_load).
    asm volatile("" : : "v"(&tile[0]) : "memory");
    const _Float16* At = &tile[ib * 4096];
    const _Float16* Bt = &tile[ib * 4096 + 2048];
#else
    {
      const int idx = tid * 8;
      const int r = idx >> 5, c = idx & 31;
      *(v8h*)&tile[r * 32 + c]        = *(const v8h*)(A + (size_t)(m0 + r) * K + kk + c);
      *(v8h*)&tile[2048 + r * 32 + c] = *(const v8h*)(W + (size_t)(n0 + r) * K + kk + c);
    }
    __syncthreads();
    const _Float16* At = &tile[0];
    const _Float16* Bt = &tile[2048];
#endif

    const _Float16* ap = &At[(mi * 16 + row) * 32 + hl * 8];
    v16h a = make_frag(ap, ap + 16);
    const int kb = hl * 8;
    const _Float16* bp0 = &Bt[((ni0 + 0) * 16 + row) * 32 + kb];
    const _Float16* bp1 = &Bt[((ni0 + 1) * 16 + row) * 32 + kb];
    v16h b0 = make_frag(bp0, bp0 + 16);
    v16h b1 = make_frag(bp1, bp1 + 16);

    acc0 = wmma_f16(a, b0, acc0);
    acc1 = wmma_f16(a, b1, acc1);
    __syncthreads();
  }

#pragma unroll
  for (int r = 0; r < 8; ++r) {
    const int m = m0 + mi * 16 + r + hl * 8;     // global row (b*S + s)
    const int bi = m >> 11;                      // /2048
    const int s  = m & (SS - 1);
#pragma unroll
    for (int t = 0; t < 2; ++t) {
      const int n = n0 + (ni0 + t) * 16 + row;   // global col
      const float v = (t == 0 ? acc0[r] : acc1[r]) + bias[n];
      if (out_mode == 2) {
        out32[(size_t)m * DD + n] = v;
      } else {
        const int h = n >> 6, dh = n & 63;
        size_t o;
        if (out_mode == 0) o = ((size_t)(bi * HH + h) * SS + s) * DHH + dh;
        else               o = ((size_t)(bi * HH + h) * DHH + dh) * SS + s;
        out16[o] = (_Float16)v;
      }
    }
  }
}

// ---------------------------------------------------------------------------
// attn_kernel: one (b, h, 16-row q tile) per block; 256 threads (8 waves).
// Dynamic LDS: 16 x 2048 fp32 logits/probs block (128 KB).
// ---------------------------------------------------------------------------
__global__ void attn_kernel(const _Float16* __restrict__ Qh,
                            const _Float16* __restrict__ Kh,
                            const _Float16* __restrict__ Vt,
                            const int* __restrict__ mask,
                            float* __restrict__ wout,
                            _Float16* __restrict__ concat) {
  extern __shared__ float P[];   // [16][SS]

  const int qt = blockIdx.x;     // 0..127
  const int h  = blockIdx.y;
  const int b  = blockIdx.z;
  const int tid = threadIdx.x;
  const int lane = tid & 31;
  const int wave = tid >> 5;
  const int row = lane & 15;
  const int hl  = lane >> 4;

  const size_t bh = (size_t)(b * HH + h);
  const _Float16* Qb = Qh + bh * SS * DHH;
  const _Float16* Kb = Kh + bh * SS * DHH;
  const _Float16* Vb = Vt + bh * DHH * SS;

  // Q fragments for this 16-row tile (reused across all k-tiles)
  const int qs = qt * 16 + row;
  const _Float16* qp = Qb + (size_t)qs * DHH;
  v16h aq0 = make_frag(qp + hl * 8,      qp + hl * 8 + 16);
  v16h aq1 = make_frag(qp + 32 + hl * 8, qp + 32 + hl * 8 + 16);

  // ---- logits: each wave owns k-tiles wave, wave+8, ... ----
  for (int kt = wave; kt < SS / 16; kt += 8) {
    const int ks = kt * 16 + row;               // B-matrix column = key index
    const _Float16* kp = Kb + (size_t)ks * DHH;
    v16h b0 = make_frag(kp + hl * 8,      kp + hl * 8 + 16);
    v16h b1 = make_frag(kp + 32 + hl * 8, kp + 32 + hl * 8 + 16);
    v8f acc = {};
    acc = wmma_f16(aq0, b0, acc);
    acc = wmma_f16(aq1, b1, acc);
#pragma unroll
    for (int r = 0; r < 8; ++r) {
      const int m = r + hl * 8;
      const int qg = qt * 16 + m;
      const int kg = kt * 16 + row;
      float v = acc[r] * 0.125f;                // 1/sqrt(DH)
      const int mk = mask[(size_t)b * SS * SS + (size_t)qg * SS + kg];
      v += (float)mk * -1.0e30f;
      P[m * SS + kg] = v;
    }
  }
  __syncthreads();

  // ---- softmax: wave w handles rows 2w, 2w+1 ----
#pragma unroll
  for (int rr = 0; rr < 2; ++rr) {
    const int m = wave * 2 + rr;
    float* prow = &P[m * SS];
    float mx = -3.0e38f;
    for (int c = lane; c < SS; c += 32) mx = fmaxf(mx, prow[c]);
#pragma unroll
    for (int off = 16; off >= 1; off >>= 1) mx = fmaxf(mx, __shfl_xor(mx, off, 32));
    float sum = 0.f;
    for (int c = lane; c < SS; c += 32) sum += __expf(prow[c] - mx);
#pragma unroll
    for (int off = 16; off >= 1; off >>= 1) sum += __shfl_xor(sum, off, 32);
    const float inv = 1.0f / sum;
    const int qg = qt * 16 + m;
    float* wrow = wout + (bh * SS + qg) * SS;
    for (int c = lane; c < SS; c += 32) {
      const float p = __expf(prow[c] - mx) * inv;
      prow[c] = p;          // keep normalized prob in LDS for PV
      wrow[c] = p;          // attention_weights output
    }
  }
  __syncthreads();

  // ---- PV: waves 0..3, wave = dh tile (nt) ----
  if (wave < 4) {
    const int nt = wave;
    v8f acc = {};
    for (int ch = 0; ch < SS / 32; ++ch) {
      // A fragment from LDS probabilities (f32 -> f16 on the fly)
      const float* pr = &P[row * SS + ch * 32 + hl * 8];
      v16h a;
#pragma unroll
      for (int j = 0; j < 8; ++j) { a[j] = (_Float16)pr[j]; a[j + 8] = (_Float16)pr[j + 16]; }
      // B fragment from V^T: contiguous along contraction (key) axis
      const _Float16* vp = Vb + (size_t)(nt * 16 + row) * SS + ch * 32 + hl * 8;
      v16h bf = make_frag(vp, vp + 16);
      acc = wmma_f16(a, bf, acc);
    }
#pragma unroll
    for (int r = 0; r < 8; ++r) {
      const int m = r + hl * 8;
      const int sg = qt * 16 + m;
      const int dh = nt * 16 + row;
      concat[((size_t)b * SS + sg) * DD + h * DHH + dh] = (_Float16)acc[r];
    }
  }
}

extern "C" void kernel_launch(void* const* d_in, const int* in_sizes, int n_in,
                              void* d_out, int out_size, void* d_ws, size_t ws_size,
                              hipStream_t stream) {
  const float* queries = (const float*)d_in[0];
  const float* keys    = (const float*)d_in[1];
  const float* values  = (const float*)d_in[2];
  const int*   mask    = (const int*)  d_in[3];
  const float* Wq = (const float*)d_in[4];  const float* bq = (const float*)d_in[5];
  const float* Wk = (const float*)d_in[6];  const float* bk = (const float*)d_in[7];
  const float* Wv = (const float*)d_in[8];  const float* bv = (const float*)d_in[9];
  const float* Wo = (const float*)d_in[10]; const float* bo = (const float*)d_in[11];

  float* out_attn = (float*)d_out;                               // (B,S,D)
  float* out_w    = out_attn + (size_t)BB * SS * DD;             // (B,H,S,S)

  const size_t nx = (size_t)BB * SS * DD;   // 4M elements (activations)
  const size_t nw = (size_t)DD * DD;        // 1M elements (weights)

  _Float16* Xq   = (_Float16*)d_ws;
  _Float16* Xk   = Xq + nx;
  _Float16* Xv   = Xk + nx;
  _Float16* Wq16 = Xv + nx;
  _Float16* Wk16 = Wq16 + nw;
  _Float16* Wv16 = Wk16 + nw;
  _Float16* Wo16 = Wv16 + nw;
  _Float16* Qh   = Wo16 + nw;
  _Float16* Kh   = Qh + nx;
  _Float16* Vt   = Kh + nx;
  _Float16* Cc   = Vt + nx;

  dim3 gblk(256);

  // f32 -> f16 pre-conversion (halves GEMM read traffic; enables TDM tiles)
  cvt_f32_to_f16<<<dim3((unsigned)(nx / 8 / 256)), gblk, 0, stream>>>(queries, Xq, (int)nx);
  cvt_f32_to_f16<<<dim3((unsigned)(nx / 8 / 256)), gblk, 0, stream>>>(keys,    Xk, (int)nx);
  cvt_f32_to_f16<<<dim3((unsigned)(nx / 8 / 256)), gblk, 0, stream>>>(values,  Xv, (int)nx);
  cvt_f32_to_f16<<<dim3((unsigned)(nw / 8 / 256)), gblk, 0, stream>>>(Wq, Wq16, (int)nw);
  cvt_f32_to_f16<<<dim3((unsigned)(nw / 8 / 256)), gblk, 0, stream>>>(Wk, Wk16, (int)nw);
  cvt_f32_to_f16<<<dim3((unsigned)(nw / 8 / 256)), gblk, 0, stream>>>(Wv, Wv16, (int)nw);
  cvt_f32_to_f16<<<dim3((unsigned)(nw / 8 / 256)), gblk, 0, stream>>>(Wo, Wo16, (int)nw);

  dim3 ggemm(BB * SS / 64, DD / 64);        // (64, 16)
  gemm16_kernel<<<ggemm, gblk, 0, stream>>>(Xq, Wq16, bq, Qh, nullptr, 0);
  gemm16_kernel<<<ggemm, gblk, 0, stream>>>(Xk, Wk16, bk, Kh, nullptr, 0);
  gemm16_kernel<<<ggemm, gblk, 0, stream>>>(Xv, Wv16, bv, Vt, nullptr, 1);

  const size_t attn_lds = (size_t)16 * SS * sizeof(float);       // 128 KB
  (void)hipFuncSetAttribute((const void*)attn_kernel,
                            hipFuncAttributeMaxDynamicSharedMemorySize, (int)attn_lds);
  dim3 gattn(SS / 16, HH, BB);              // (128, 16, 2)
  attn_kernel<<<gattn, gblk, attn_lds, stream>>>(Qh, Kh, Vt, mask, out_w, Cc);

  gemm16_kernel<<<ggemm, gblk, 0, stream>>>(Cc, Wo16, bo, nullptr, out_attn, 2);
}
